// Prediction_652835029358
// MI455X (gfx1250) — matrix operand order repaired and checked by reading, not compiled, not run on hardware
//
#include <hip/hip_runtime.h>
#include <hip/hip_bf16.h>
#include <stdint.h>

// Problem constants (match reference)
constexpr int B_ = 64, S_ = 512, H_ = 1024, NC_ = 20, NN_ = 15, OP_ = 5;
constexpr int NCN_ = NC_ + NN_;
#define NEGV (-1e12f)

typedef __attribute__((ext_vector_type(16))) _Float16 v16h;
typedef __attribute__((ext_vector_type(8)))  _Float16 v8h;
typedef __attribute__((ext_vector_type(8)))  float    v8f;

// Build a 16-half WMMA fragment from two 16-byte-aligned groups of 8 halves.
__device__ __forceinline__ v16h frag_from(const _Float16* p0, const _Float16* p1) {
  v8h x = *(const v8h*)p0;
  v8h y = *(const v8h*)p1;
  v16h r;
#pragma unroll
  for (int i = 0; i < 8; ++i) { r[i] = x[i]; r[i + 8] = y[i]; }
  return r;
}

// ---------------------------------------------------------------------------
// Generic WMMA GEMM: C[M,N] = A[M,K](f16) @ W[N,K]^T(f16) + bias, opt tanh.
// Block = 128 threads (4 waves). Each wave computes a 16x64 strip: one A
// fragment is reused across 4 N-tile accumulators (4 wmma per A load).
// A frag per ISA 16-bit A layout: half j <-> K = (j/8)*16 + hi*8 + (j%8).
// B frag: lane column n; halves take the same K order as A, from a W row (W is [N,K]).
// ---------------------------------------------------------------------------
__global__ void k_wmma_gemm(const _Float16* __restrict__ A, const _Float16* __restrict__ W,
                            const float* __restrict__ bias, float* __restrict__ C,
                            int M, int N, int K, int act) {
  const int wave = threadIdx.x >> 5;
  const int lane = threadIdx.x & 31;
  const int hi = lane >> 4, nl = lane & 15;
  const int n_base = blockIdx.x * 256 + wave * 64;   // 4 consecutive N-tiles
  const int m0 = blockIdx.y * 16;
  if (n_base >= N || m0 >= M) return;
  const _Float16* arow = A + (size_t)(m0 + nl) * K;
  const _Float16* wrow[4];
#pragma unroll
  for (int j = 0; j < 4; ++j) wrow[j] = W + (size_t)(n_base + j * 16 + nl) * K;
  v8f c[4];
#pragma unroll
  for (int j = 0; j < 4; ++j) c[j] = (v8f){};
  for (int k = 0; k < K; k += 32) {
    v16h a = frag_from(arow + k + hi * 8, arow + k + 16 + hi * 8);
#pragma unroll
    for (int j = 0; j < 4; ++j) {
      v16h b = frag_from(wrow[j] + k + hi * 8, wrow[j] + k + 16 + hi * 8);
      c[j] = __builtin_amdgcn_wmma_f32_16x16x32_f16(false, a, false, b, (short)0, c[j], false, false);
    }
  }
#pragma unroll
  for (int j = 0; j < 4; ++j) {
    const int n = n_base + j * 16 + nl;
    const float bv = bias ? bias[n] : 0.0f;
#pragma unroll
    for (int r = 0; r < 8; ++r) {
      const int m = m0 + hi * 8 + r;  // C/D layout: VGPR r = row hi*8+r, col = lane&15
      float v = c[j][r] + bv;
      if (act == 1) v = tanhf(v);
      C[(size_t)m * N + n] = v;
    }
  }
}

// ---------------------------------------------------------------------------
// Fused score head:
//   out[row] = bias_s + sum_n vvec[n] * tanh((A@W^T)[row,n] + addv[row/rpb, n])
// MT rows per block staged in LDS. fp32 source is converted on the fly
// (attention path); f16 source is staged via CDNA5 async global->LDS DMA
// (global_load_async_to_lds_b128, ASYNCcnt). 4 waves x 16 N-tiles each,
// processed in groups of 4 with MT/16 M-subtiles -> A and B fragment reuse.
// Per-row partials reduced via shfl-xor + LDS float atomics.
// ---------------------------------------------------------------------------
template <int MT>
__global__ void k_wmma_fused_score(const void* __restrict__ Asrc, int a_is_f16,
                                   const _Float16* __restrict__ W,
                                   const float* __restrict__ addv,
                                   const float* __restrict__ vvec,
                                   const float* __restrict__ bias_s,
                                   float* __restrict__ out,
                                   int K, int rows_per_b) {
  extern __shared__ __align__(16) char smem_raw[];
  _Float16* As = (_Float16*)smem_raw;                      // [MT, K]
  float* outAcc = (float*)(smem_raw + (size_t)MT * K * 2); // [MT]
  constexpr int MSUB = MT / 16;
  const int tid = threadIdx.x;
  const int lane = tid & 31, wave = tid >> 5;
  const int hi = lane >> 4, nl = lane & 15;
  const int m0 = blockIdx.x * MT;

  if (a_is_f16) {
    // Async DMA stage: 16B chunks, bypasses VGPRs, tracked on ASYNCcnt.
    const _Float16* Ag = (const _Float16*)Asrc + (size_t)m0 * K;
    const int chunks = MT * K * 2 / 16;
    for (int i = tid; i < chunks; i += blockDim.x) {
      uint32_t laddr = (uint32_t)(uintptr_t)(As) + (uint32_t)i * 16u;
      uint64_t gaddr = (uint64_t)(uintptr_t)Ag + (uint64_t)i * 16u;
      asm volatile("global_load_async_to_lds_b128 %0, %1, off"
                   :: "v"(laddr), "v"(gaddr) : "memory");
    }
    asm volatile("s_wait_asynccnt 0x0" ::: "memory");
  } else {
    const float* Af = (const float*)Asrc + (size_t)m0 * K;
    for (int i = tid; i < MT * K; i += blockDim.x) As[i] = (_Float16)Af[i];
  }
  if (tid < MT) outAcc[tid] = 0.0f;
  __syncthreads();

  float part[MSUB][8];
#pragma unroll
  for (int ms = 0; ms < MSUB; ++ms)
#pragma unroll
    for (int r = 0; r < 8; ++r) part[ms][r] = 0.0f;

  // Wave handles N-tiles [wave*16, wave*16+16), in 4 groups of 4.
  for (int ng = 0; ng < 4; ++ng) {
    const int t0 = wave * 16 + ng * 4;
    const _Float16* wrow[4];
#pragma unroll
    for (int j = 0; j < 4; ++j) wrow[j] = W + (size_t)((t0 + j) * 16 + nl) * K;
    v8f c[MSUB][4];
#pragma unroll
    for (int ms = 0; ms < MSUB; ++ms)
#pragma unroll
      for (int j = 0; j < 4; ++j) c[ms][j] = (v8f){};
    for (int k = 0; k < K; k += 32) {
      v16h a[MSUB];
#pragma unroll
      for (int ms = 0; ms < MSUB; ++ms) {
        const _Float16* ap = As + (size_t)(ms * 16 + nl) * K + k;
        a[ms] = frag_from(ap + hi * 8, ap + 16 + hi * 8);
      }
#pragma unroll
      for (int j = 0; j < 4; ++j) {
        v16h b = frag_from(wrow[j] + k + hi * 8, wrow[j] + k + 16 + hi * 8);
#pragma unroll
        for (int ms = 0; ms < MSUB; ++ms)
          c[ms][j] = __builtin_amdgcn_wmma_f32_16x16x32_f16(false, a[ms], false, b, (short)0, c[ms][j], false, false);
      }
    }
#pragma unroll
    for (int j = 0; j < 4; ++j) {
      const int n = (t0 + j) * 16 + nl;
      const float vn = vvec[n];
#pragma unroll
      for (int ms = 0; ms < MSUB; ++ms) {
#pragma unroll
        for (int r = 0; r < 8; ++r) {
          const int grow = m0 + ms * 16 + hi * 8 + r;
          const int bb = grow / rows_per_b;
          part[ms][r] += vn * tanhf(c[ms][j][r] + addv[(size_t)bb * H_ + n]);
        }
      }
    }
  }
  // Reduce over the 16 lanes (N columns) of each half-wave.
#pragma unroll
  for (int off = 1; off < 16; off <<= 1)
#pragma unroll
    for (int ms = 0; ms < MSUB; ++ms)
#pragma unroll
      for (int r = 0; r < 8; ++r) part[ms][r] += __shfl_xor(part[ms][r], off, 32);
  if (nl == 0) {
#pragma unroll
    for (int ms = 0; ms < MSUB; ++ms)
#pragma unroll
      for (int r = 0; r < 8; ++r) atomicAdd(&outAcc[ms * 16 + hi * 8 + r], part[ms][r]);
  }
  __syncthreads();
  if (tid < MT) {
    const float bsv = bias_s ? bias_s[0] : 0.0f;
    out[m0 + tid] = bsv + outAcc[tid];
  }
}

// ---------------------------------------------------------------------------
// Elementwise / helper kernels
// ---------------------------------------------------------------------------
__global__ void k_f32_to_f16(_Float16* __restrict__ dst, int dstld,
                             const float* __restrict__ src, int srcld,
                             int rows, int cols) {
  int i = blockIdx.x * blockDim.x + threadIdx.x;
  if (i >= rows * cols) return;
  int r = i / cols, c = i % cols;
  dst[(size_t)r * dstld + c] = (_Float16)src[(size_t)r * srcld + c];
}

__global__ void k_node(const float* __restrict__ tl, const float* __restrict__ tlg,
                       const float* __restrict__ tr, const float* __restrict__ trg,
                       const unsigned char* __restrict__ left_mask,
                       float* __restrict__ node_out, _Float16* __restrict__ node16,
                       _Float16* __restrict__ nodectx16) {
  int i = blockIdx.x * blockDim.x + threadIdx.x;
  if (i >= B_ * H_) return;
  int b = i / H_, h = i % H_;
  float v;
  if (left_mask[b]) v = tanhf(tr[i]) * (1.0f / (1.0f + expf(-trg[i])));
  else              v = tanhf(tl[i]) * (1.0f / (1.0f + expf(-tlg[i])));
  node_out[i] = v;
  node16[i] = (_Float16)v;
  nodectx16[(size_t)b * 2 * H_ + h] = (_Float16)v;
}

__global__ void k_softmax(const float* __restrict__ energies, const int* __restrict__ seq_mask,
                          float* __restrict__ attn) {
  __shared__ float red[256];
  const int b = blockIdx.x, tid = threadIdx.x;
  float mx = -3.4e38f;
  for (int s = tid; s < S_; s += 256) {
    float e = seq_mask[b * S_ + s] == 0 ? NEGV : energies[b * S_ + s];
    mx = fmaxf(mx, e);
  }
  red[tid] = mx; __syncthreads();
  for (int o = 128; o > 0; o >>= 1) { if (tid < o) red[tid] = fmaxf(red[tid], red[tid + o]); __syncthreads(); }
  mx = red[0]; __syncthreads();
  float sum = 0.0f;
  for (int s = tid; s < S_; s += 256) {
    float e = seq_mask[b * S_ + s] == 0 ? NEGV : energies[b * S_ + s];
    sum += expf(e - mx);
  }
  red[tid] = sum; __syncthreads();
  for (int o = 128; o > 0; o >>= 1) { if (tid < o) red[tid] += red[tid + o]; __syncthreads(); }
  const float inv = 1.0f / red[0];
  for (int s = tid; s < S_; s += 256) {
    float e = seq_mask[b * S_ + s] == 0 ? NEGV : energies[b * S_ + s];
    attn[b * S_ + s] = expf(e - mx) * inv;
  }
}

__global__ void k_ctx(const float* __restrict__ attn, const float* __restrict__ enc,
                      float* __restrict__ ctx_out, _Float16* __restrict__ nodectx16) {
  __shared__ float sAttn[S_];
  const int b = blockIdx.x;
  for (int s = threadIdx.x; s < S_; s += blockDim.x) sAttn[s] = attn[b * S_ + s];
  __syncthreads();
  for (int h = threadIdx.x; h < H_; h += blockDim.x) {
    float acc = 0.0f;
    const float* ep = enc + (size_t)b * S_ * H_ + h;
    for (int s = 0; s < S_; ++s) acc += sAttn[s] * ep[(size_t)s * H_];
    ctx_out[b * H_ + h] = acc;
    nodectx16[(size_t)b * 2 * H_ + H_ + h] = (_Float16)acc;
  }
}

__global__ void k_build_cn(const float* __restrict__ const_embed, const float* __restrict__ num_pades,
                           float* __restrict__ cn) {
  int i = blockIdx.x * blockDim.x + threadIdx.x;
  if (i >= B_ * NCN_ * H_) return;
  int h = i % H_; int t = i / H_; int n = t % NCN_; int b = t / NCN_;
  cn[i] = (n < NC_) ? const_embed[n * H_ + h]
                    : num_pades[((size_t)b * NN_ + (n - NC_)) * H_ + h];
}

// dst[row] = [ emb(b,n) | emb(b,n)*leaf(b) ]  as f16, row-major [M, 2H]
__global__ void k_build_scoreA(_Float16* __restrict__ dst, const float* __restrict__ emb,
                               const float* __restrict__ leaf, int M, int epb, long ebs) {
  int i = blockIdx.x * blockDim.x + threadIdx.x;
  if (i >= M * H_) return;
  int h = i % H_; int row = i / H_;
  int b = row / epb, n = row % epb;
  float e = emb[(size_t)b * ebs + (size_t)n * H_ + h];
  float l = leaf[(size_t)b * H_ + h];
  dst[(size_t)row * 2 * H_ + h] = (_Float16)e;
  dst[(size_t)row * 2 * H_ + H_ + h] = (_Float16)(e * l);
}

__global__ void k_mask_num(const float* __restrict__ raw, const int* __restrict__ mask_nums,
                           float* __restrict__ out) {
  int i = blockIdx.x * blockDim.x + threadIdx.x;
  if (i >= B_ * NCN_) return;
  out[i] = mask_nums[i] == 0 ? NEGV : raw[i];
}

__global__ void k_copy_f32(const float* __restrict__ src, float* __restrict__ dst, int n) {
  int i = blockIdx.x * blockDim.x + threadIdx.x;
  if (i < n) dst[i] = src[i];
}

// ---------------------------------------------------------------------------
extern "C" void kernel_launch(void* const* d_in, const int* in_sizes, int n_in,
                              void* d_out, int out_size, void* d_ws, size_t ws_size,
                              hipStream_t stream) {
  const float* cur_emb     = (const float*)d_in[0];
  const float* left_childs = (const float*)d_in[1];
  const unsigned char* left_mask = (const unsigned char*)d_in[2];  // bool array
  const float* enc         = (const float*)d_in[3];
  const float* num_pades   = (const float*)d_in[4];
  const int*   seq_mask    = (const int*)d_in[5];
  const int*   mask_nums   = (const int*)d_in[6];
  const float* const_embed = (const float*)d_in[7];
  const float* op_embed    = (const float*)d_in[8];
  const float *Wl=(const float*)d_in[9],  *bl=(const float*)d_in[10];
  const float *Wlg=(const float*)d_in[11],*blg=(const float*)d_in[12];
  const float *Wr=(const float*)d_in[13], *br=(const float*)d_in[14];
  const float *Wrg=(const float*)d_in[15],*brg=(const float*)d_in[16];
  const float *Wleaf=(const float*)d_in[17],*bleaf=(const float*)d_in[18];
  const float *Wa=(const float*)d_in[19], *ba=(const float*)d_in[20];
  const float *Ws=(const float*)d_in[21], *bs=(const float*)d_in[22];
  const float *Wna=(const float*)d_in[23],*bna=(const float*)d_in[24];
  const float *Wns=(const float*)d_in[25];
  const float *Woa=(const float*)d_in[26],*boa=(const float*)d_in[27];
  const float *Wos=(const float*)d_in[28];

  float* out = (float*)d_out;
  float* o_num  = out;                 // [B,35]
  float* o_op   = out + 2240;          // [B,5]
  float* o_node = out + 2560;          // [B,1,H]
  float* o_ctx  = out + 68096;         // [B,1,H]
  float* o_cn   = out + 133632;        // [B,35,H]
  float* o_ope  = out + 2427392;       // [5,H]

  // Workspace carve-out (bump allocator, 256B aligned)
  char* p = (char*)d_ws;
  auto alloc = [&](size_t bytes) { char* r = p; p += (bytes + 255) & ~(size_t)255; return r; };
  _Float16* Wl16     = (_Float16*)alloc((size_t)H_*H_*2);
  _Float16* Wlg16    = (_Float16*)alloc((size_t)H_*H_*2);
  _Float16* Wr16     = (_Float16*)alloc((size_t)H_*2*H_*2);
  _Float16* Wrg16    = (_Float16*)alloc((size_t)H_*2*H_*2);
  _Float16* WaN16    = (_Float16*)alloc((size_t)H_*H_*2);     // Wa[:, :H]
  _Float16* WaE16    = (_Float16*)alloc((size_t)H_*H_*2);     // Wa[:, H:]
  _Float16* Wleaf16  = (_Float16*)alloc((size_t)H_*2*H_*2);
  _Float16* Wna1_16  = (_Float16*)alloc((size_t)H_*H_*2);     // Wna[:, :H]
  _Float16* Wna23_16 = (_Float16*)alloc((size_t)H_*2*H_*2);   // [Wna[:,H:2H] | Wna[:,2H:]]
  _Float16* Woa1_16  = (_Float16*)alloc((size_t)H_*H_*2);
  _Float16* Woa23_16 = (_Float16*)alloc((size_t)H_*2*H_*2);
  _Float16* cur16    = (_Float16*)alloc((size_t)B_*H_*2);
  _Float16* lc16     = (_Float16*)alloc((size_t)B_*2*H_*2);
  _Float16* node16   = (_Float16*)alloc((size_t)B_*H_*2);
  _Float16* nodectx16= (_Float16*)alloc((size_t)B_*2*H_*2);
  _Float16* leaf16   = (_Float16*)alloc((size_t)B_*H_*2);
  _Float16* Anum16   = (_Float16*)alloc((size_t)B_*NCN_*2*H_*2);
  _Float16* Aop16    = (_Float16*)alloc((size_t)B_*OP_*2*H_*2);
  float* t_l      = (float*)alloc((size_t)B_*H_*4);
  float* t_lg     = (float*)alloc((size_t)B_*H_*4);
  float* t_r      = (float*)alloc((size_t)B_*H_*4);
  float* t_rg     = (float*)alloc((size_t)B_*H_*4);
  float* pre_node = (float*)alloc((size_t)B_*H_*4);
  float* energies = (float*)alloc((size_t)B_*S_*4);
  float* attn     = (float*)alloc((size_t)B_*S_*4);
  float* leaf_f32 = (float*)alloc((size_t)B_*H_*4);
  float* pna1     = (float*)alloc((size_t)B_*H_*4);
  float* poa1     = (float*)alloc((size_t)B_*H_*4);
  float* num_raw  = (float*)alloc((size_t)B_*NCN_*4);

  const int TB = 256;
  auto g1 = [&](int n) { return (n + TB - 1) / TB; };

  // ---- Stage 1: fp32 -> f16 conversions (weights + small activations)
  k_f32_to_f16<<<g1(H_*H_), TB, 0, stream>>>(Wl16, H_, Wl, H_, H_, H_);
  k_f32_to_f16<<<g1(H_*H_), TB, 0, stream>>>(Wlg16, H_, Wlg, H_, H_, H_);
  k_f32_to_f16<<<g1(H_*2*H_), TB, 0, stream>>>(Wr16, 2*H_, Wr, 2*H_, H_, 2*H_);
  k_f32_to_f16<<<g1(H_*2*H_), TB, 0, stream>>>(Wrg16, 2*H_, Wrg, 2*H_, H_, 2*H_);
  k_f32_to_f16<<<g1(H_*H_), TB, 0, stream>>>(WaN16, H_, Wa,       2*H_, H_, H_);
  k_f32_to_f16<<<g1(H_*H_), TB, 0, stream>>>(WaE16, H_, Wa + H_,  2*H_, H_, H_);
  k_f32_to_f16<<<g1(H_*2*H_), TB, 0, stream>>>(Wleaf16, 2*H_, Wleaf, 2*H_, H_, 2*H_);
  k_f32_to_f16<<<g1(H_*H_), TB, 0, stream>>>(Wna1_16, H_, Wna,        3*H_, H_, H_);
  k_f32_to_f16<<<g1(H_*H_), TB, 0, stream>>>(Wna23_16,       2*H_, Wna + H_,   3*H_, H_, H_);
  k_f32_to_f16<<<g1(H_*H_), TB, 0, stream>>>(Wna23_16 + H_,  2*H_, Wna + 2*H_, 3*H_, H_, H_);
  k_f32_to_f16<<<g1(H_*H_), TB, 0, stream>>>(Woa1_16, H_, Woa,        3*H_, H_, H_);
  k_f32_to_f16<<<g1(H_*H_), TB, 0, stream>>>(Woa23_16,       2*H_, Woa + H_,   3*H_, H_, H_);
  k_f32_to_f16<<<g1(H_*H_), TB, 0, stream>>>(Woa23_16 + H_,  2*H_, Woa + 2*H_, 3*H_, H_, H_);
  k_f32_to_f16<<<g1(B_*H_), TB, 0, stream>>>(cur16, H_, cur_emb, H_, B_, H_);
  k_f32_to_f16<<<g1(B_*H_), TB, 0, stream>>>(lc16,      2*H_, left_childs, H_, B_, H_);
  k_f32_to_f16<<<g1(B_*H_), TB, 0, stream>>>(lc16 + H_, 2*H_, cur_emb,     H_, B_, H_);

  // ---- Stage 2: gate GEMMs + node combine
  dim3 blk(128);
  dim3 gS(H_ / 256, B_ / 16);  // N=1024 -> 4 x-blocks (64 cols/wave); M=64 -> 4 y-blocks
  k_wmma_gemm<<<gS, blk, 0, stream>>>(cur16, Wl16,  bl,  t_l,  B_, H_, H_,   0);
  k_wmma_gemm<<<gS, blk, 0, stream>>>(cur16, Wlg16, blg, t_lg, B_, H_, H_,   0);
  k_wmma_gemm<<<gS, blk, 0, stream>>>(lc16,  Wr16,  br,  t_r,  B_, H_, 2*H_, 0);
  k_wmma_gemm<<<gS, blk, 0, stream>>>(lc16,  Wrg16, brg, t_rg, B_, H_, 2*H_, 0);
  k_node<<<g1(B_*H_), TB, 0, stream>>>(t_l, t_lg, t_r, t_rg, left_mask, o_node, node16, nodectx16);

  // ---- Stage 3: attention (node part small GEMM, then fused enc GEMM->tanh->dot)
  k_wmma_gemm<<<gS, blk, 0, stream>>>(node16, WaN16, ba, pre_node, B_, H_, H_, 0);
  {
    size_t shmem = (size_t)32 * H_ * 2 + 32 * 4;  // 32-row A tile + accumulators
    k_wmma_fused_score<32><<<B_ * S_ / 32, blk, shmem, stream>>>(
        (const void*)enc, 0, WaE16, pre_node, Ws, bs, energies, H_, S_);
  }
  k_softmax<<<B_, TB, 0, stream>>>(energies, seq_mask, attn);
  k_ctx<<<B_, TB, 0, stream>>>(attn, enc, o_ctx, nodectx16);

  // ---- Stage 4: leaf = tanh([node|ctx] @ Wleaf^T + bleaf)
  k_wmma_gemm<<<gS, blk, 0, stream>>>(nodectx16, Wleaf16, bleaf, leaf_f32, B_, H_, 2*H_, 1);
  k_f32_to_f16<<<g1(B_*H_), TB, 0, stream>>>(leaf16, H_, leaf_f32, H_, B_, H_);

  // ---- Stage 5: score heads. leaf-part GEMMs (include bias), then fused score.
  k_wmma_gemm<<<gS, blk, 0, stream>>>(leaf16, Wna1_16, bna, pna1, B_, H_, H_, 0);
  k_wmma_gemm<<<gS, blk, 0, stream>>>(leaf16, Woa1_16, boa, poa1, B_, H_, H_, 0);
  k_build_cn<<<g1(B_*NCN_*H_), TB, 0, stream>>>(const_embed, num_pades, o_cn);
  k_build_scoreA<<<g1(B_*NCN_*H_), TB, 0, stream>>>(Anum16, o_cn, leaf_f32, B_*NCN_, NCN_, (long)NCN_*H_);
  k_build_scoreA<<<g1(B_*OP_*H_),  TB, 0, stream>>>(Aop16, op_embed, leaf_f32, B_*OP_, OP_, 0L);
  {
    size_t shmem = (size_t)16 * 2 * H_ * 2 + 16 * 4;  // 16-row A tile (K=2H)
    k_wmma_fused_score<16><<<B_ * NCN_ / 16, blk, shmem, stream>>>(
        (const void*)Anum16, 1, Wna23_16, pna1, Wns, nullptr, num_raw, 2*H_, NCN_);
    k_wmma_fused_score<16><<<B_ * OP_ / 16, blk, shmem, stream>>>(
        (const void*)Aop16, 1, Woa23_16, poa1, Wos, nullptr, o_op, 2*H_, OP_);
  }
  k_mask_num<<<g1(B_*NCN_), TB, 0, stream>>>(num_raw, mask_nums, o_num);

  // ---- Stage 6: passthrough output (op_embed)
  k_copy_f32<<<g1(OP_*H_), TB, 0, stream>>>(op_embed, o_ope, OP_ * H_);
}
